// PairNN_51238959841773
// MI455X (gfx1250) — matrix operand order repaired and checked by reading, not compiled, not run on hardware
//
#include <hip/hip_runtime.h>
#include <hip/hip_bf16.h>
#include <math.h>

#define NATOMS 2048
#define KNEI   48
#define NPAIRS (NATOMS*KNEI)
#define HID    64
#define PI_F   3.14159265358979323846f

typedef __attribute__((ext_vector_type(16))) _Float16 v16h;
typedef __attribute__((ext_vector_type(8)))  float    v8f;

__device__ __forceinline__ v8f wmma_f16(v16h a, v16h b, v8f c) {
  // (neg_a, A, neg_b, B, c_mod, C, reuse_a, reuse_b)
  return __builtin_amdgcn_wmma_f32_16x16x32_f16(false, a, false, b, (short)0, c, false, false);
}

// Branch-free tanh: v_exp_f32 + v_rcp_f32, no EXEC save/restore around WMMAs.
__device__ __forceinline__ float fast_tanh(float x) {
  float ax = fabsf(x);
  float e  = __expf(-2.0f * ax);
  float t  = (1.0f - e) * __builtin_amdgcn_rcpf(1.0f + e);
  return copysignf(t, x);
}

// -------------------------------------------------------------------------
// K1: per-pair geometry + radial basis (and its d-derivatives)
// wsA[p*8]  = {d, dnx, dny, dnz, fc, dfc/dd, fcrik, dfcrik/dd}
// wsB[p*12] = {B0..B4, B0'..B4', pad, pad}   (rbf_n = B_n * fc)
// desc[p*32]: [0..4]=rbf, [5..16]=g3b (filled by K2), [17..31]=0
// -------------------------------------------------------------------------
__global__ void pair_prep_kernel(const float* __restrict__ rij,
                                 float* __restrict__ wsA,
                                 float* __restrict__ wsB,
                                 float* __restrict__ desc) {
  int p = blockIdx.x * blockDim.x + threadIdx.x;
  if (p >= NPAIRS) return;
  float x = rij[3*p+0], y = rij[3*p+1], z = rij[3*p+2];
  float d  = sqrtf(x*x + y*y + z*z);
  float dm = fmaxf(d, 1e-12f);
  float invm = 1.0f / dm;
  float dnx = x*invm, dny = y*invm, dnz = z*invm;

  // pair cutoff: t = (d-RMIN)/(CUT-RMIN), RMIN=3.5, CUT=3.0 -> 1/(CUT-RMIN) = -2
  float fc, dfc;
  if (d > 3.5f) {
    float tt = (d - 3.5f) * (-2.0f);
    fc  = 0.5f + 0.5f * cosf(PI_F * tt);
    dfc = PI_F * sinf(PI_F * tt);            // -0.5*pi*sin(pi t)*(-2)
  } else { fc = 1.0f; dfc = 0.0f; }

  float ar   = PI_F * d / 3.0f;
  float fcr  = 0.5f + 0.5f * cosf(ar);
  float dfcr = -0.5f * (PI_F / 3.0f) * sinf(ar);

  float* A = wsA + (size_t)p * 8;
  A[0]=d; A[1]=dnx; A[2]=dny; A[3]=dnz; A[4]=fc; A[5]=dfc; A[6]=fcr; A[7]=dfcr;

  float invd = 1.0f / d;                      // reference divides by raw d
  const float s23 = 0.81649658092f;           // sqrt(2/3)
  float* B = wsB  + (size_t)p * 12;
  float* D = desc + (size_t)p * 32;
  #pragma unroll
  for (int n = 0; n < 5; ++n) {
    float a  = (float)(n + 1) * (PI_F / 3.0f);
    float sn = sinf(a * d), cn = cosf(a * d);
    float Bn  = s23 * sn * invd;
    float Bpn = s23 * (a * cn * invd - sn * invd * invd);
    B[n] = Bn; B[5+n] = Bpn;
    D[n] = Bn * fc;
  }
  #pragma unroll
  for (int f = 17; f < 32; ++f) D[f] = 0.0f;  // zero K-padding for WMMA A operand
}

// -------------------------------------------------------------------------
// K2: per-atom 3-body forward. cosang in LDS, g3b -> desc[5..16]
// -------------------------------------------------------------------------
__global__ __launch_bounds__(64) void threebody_fwd_kernel(const float* __restrict__ wsA,
                                                           float* __restrict__ desc) {
  __shared__ float sdn[KNEI*3];
  __shared__ float sw[KNEI];
  __shared__ float scos[KNEI*KNEI];
  int t = threadIdx.x;
  int atom = blockIdx.x;
  if (t < KNEI) {
    const float* A = wsA + ((size_t)atom*KNEI + t) * 8;
    sdn[t*3+0]=A[1]; sdn[t*3+1]=A[2]; sdn[t*3+2]=A[3];
    sw[t] = A[6];
  }
  __syncthreads();
  for (int e = t; e < KNEI*KNEI; e += 64) {
    int k = e / KNEI, l = e - k*KNEI;
    float c = (k == l) ? 0.0f
            : (sdn[k*3]*sdn[l*3] + sdn[k*3+1]*sdn[l*3+1] + sdn[k*3+2]*sdn[l*3+2]);
    scos[e] = c;
  }
  __syncthreads();
  if (t < KNEI) {
    float acc[12];
    #pragma unroll
    for (int m = 0; m < 12; ++m) acc[m] = 0.0f;
    const float* row = scos + t*KNEI;
    for (int l = 0; l < KNEI; ++l) {
      float c = row[l], w = sw[l];
      #pragma unroll
      for (int m = 0; m < 12; ++m) {
        float u = c - (-1.0f + 0.18181818181818182f * (float)m);  // mu_m
        acc[m] += w * __expf(-4.0f * u * u);                      // ETA=4
      }
    }
    float* D = desc + ((size_t)atom*KNEI + t) * 32 + 5;
    #pragma unroll
    for (int m = 0; m < 12; ++m) D[m] = acc[m];
  }
}

// -------------------------------------------------------------------------
// K3: WMMA MLP forward + backward for a tile of 16 pairs (one wave32/block).
// Writes eij[p] and gdesc[p*32 + 0..16].
// Layouts used here (wave32; C/D as in CDNA5 ISA 7.12.2):
//   A 16x32 f16: M=lane&15; elems 0..7 -> K = (lane<16?0:16)+j; 8..15 -> K=(lane<16?8:24)+(j-8)
//   B 32x16 f16: N=lane&15; elem j -> K = (lane<16?0:16)+j (contiguous, this code's choice, same as A)
//   C/D 16x16 f32: vgpr r -> M = r + (lane<16?0:8), N = lane&15
// -------------------------------------------------------------------------
__global__ __launch_bounds__(32) void mlp_fwd_bwd_kernel(
    const float* __restrict__ desc, const float* __restrict__ wsA,
    const float* __restrict__ W1, const float* __restrict__ b1,
    const float* __restrict__ W2, const float* __restrict__ b2,
    const float* __restrict__ W3, const float* __restrict__ b3,
    float* __restrict__ gdesc, float* __restrict__ eout) {
  __shared__ float h1s[16*HID];
  __shared__ float h2s[16*HID];
  __shared__ float gas[16*HID];
  __shared__ float gbs[16*HID];
  __shared__ float fcs[16];

  const int lane  = threadIdx.x;
  const int m     = lane & 15;
  const int hi    = lane >> 4;
  const int tile  = blockIdx.x;
  const int rbase = hi * 8;
  const int kb0   = hi * 16;
  const int kb1   = hi * 16 + 8;
  const int kbB   = hi * 16;
  const size_t pair_m = (size_t)tile * 16 + m;

  if (lane < 16) fcs[m] = wsA[pair_m*8 + 4];

  // ---- layer 1: h1 = tanh(desc @ W1 + b1), K padded 17->32 ----
  v16h a0;
  {
    const float* dp = desc + pair_m * 32;
    #pragma unroll
    for (int j = 0; j < 8; ++j) {
      a0[j]   = (_Float16)dp[kb0 + j];
      a0[8+j] = (_Float16)dp[kb1 + j];
    }
  }
  #pragma unroll
  for (int n = 0; n < 4; ++n) {
    const int col = n*16 + m;
    v16h b;
    #pragma unroll
    for (int j = 0; j < 16; ++j) {
      int kk = kbB + j;
      b[j] = (_Float16)((kk < 17) ? W1[kk*HID + col] : 0.0f);
    }
    v8f c; float bias = b1[col];
    #pragma unroll
    for (int r = 0; r < 8; ++r) c[r] = bias;
    c = wmma_f16(a0, b, c);
    #pragma unroll
    for (int r = 0; r < 8; ++r) h1s[(rbase + r)*HID + col] = fast_tanh(c[r]);
  }
  __syncthreads();

  // ---- layer 2: h2 = tanh(h1 @ W2 + b2), K=64 split in two WMMAs ----
  v16h aL, aH;
  #pragma unroll
  for (int j = 0; j < 8; ++j) {
    aL[j]   = (_Float16)h1s[m*HID + kb0 + j];
    aL[8+j] = (_Float16)h1s[m*HID + kb1 + j];
    aH[j]   = (_Float16)h1s[m*HID + 32 + kb0 + j];
    aH[8+j] = (_Float16)h1s[m*HID + 32 + kb1 + j];
  }
  #pragma unroll
  for (int n = 0; n < 4; ++n) {
    const int col = n*16 + m;
    v16h bL, bH;
    #pragma unroll
    for (int j = 0; j < 16; ++j) {
      bL[j] = (_Float16)W2[(kbB + j)*HID + col];
      bH[j] = (_Float16)W2[(32 + kbB + j)*HID + col];
    }
    v8f c; float bias = b2[col];
    #pragma unroll
    for (int r = 0; r < 8; ++r) c[r] = bias;
    c = wmma_f16(aL, bL, c);
    c = wmma_f16(aH, bH, c);
    #pragma unroll
    for (int r = 0; r < 8; ++r) h2s[(rbase + r)*HID + col] = fast_tanh(c[r]);
  }
  __syncthreads();

  // ---- layer 3: eij = h2 @ W3 + b3 (tiny, VALU) ----
  if (lane < 16) {
    float e = b3[0];
    for (int kk = 0; kk < HID; ++kk) e += h2s[m*HID + kk] * W3[kk];
    eout[pair_m] = e;
  }

  // ---- backward: g_eij = fc ; g_a2 = fc * W3 * (1 - h2^2) ----
  for (int idx = lane; idx < 16*HID; idx += 32) {
    int mm = idx >> 6, kk = idx & 63;
    float h = h2s[idx];
    gas[idx] = fcs[mm] * W3[kk] * (1.0f - h*h);
  }
  __syncthreads();

  // ---- g_h1 = g_a2 @ W2^T ; g_a1 = g_h1 * (1 - h1^2) ----
  v16h gL, gH;
  #pragma unroll
  for (int j = 0; j < 8; ++j) {
    gL[j]   = (_Float16)gas[m*HID + kb0 + j];
    gL[8+j] = (_Float16)gas[m*HID + kb1 + j];
    gH[j]   = (_Float16)gas[m*HID + 32 + kb0 + j];
    gH[8+j] = (_Float16)gas[m*HID + 32 + kb1 + j];
  }
  #pragma unroll
  for (int n = 0; n < 4; ++n) {
    const int col = n*16 + m;
    v16h bL, bH;                                  // B[k][o] = W2[o][k]
    #pragma unroll
    for (int j = 0; j < 16; ++j) {
      bL[j] = (_Float16)W2[col*HID + (kbB + j)];
      bH[j] = (_Float16)W2[col*HID + (32 + kbB + j)];
    }
    v8f c;
    #pragma unroll
    for (int r = 0; r < 8; ++r) c[r] = 0.0f;
    c = wmma_f16(gL, bL, c);
    c = wmma_f16(gH, bH, c);
    #pragma unroll
    for (int r = 0; r < 8; ++r) {
      float h = h1s[(rbase + r)*HID + col];
      gbs[(rbase + r)*HID + col] = c[r] * (1.0f - h*h);
    }
  }
  __syncthreads();

  // ---- g_desc = g_a1 @ W1^T  (N padded 17->32) ----
  v16h qL, qH;
  #pragma unroll
  for (int j = 0; j < 8; ++j) {
    qL[j]   = (_Float16)gbs[m*HID + kb0 + j];
    qL[8+j] = (_Float16)gbs[m*HID + kb1 + j];
    qH[j]   = (_Float16)gbs[m*HID + 32 + kb0 + j];
    qH[8+j] = (_Float16)gbs[m*HID + 32 + kb1 + j];
  }
  #pragma unroll
  for (int n = 0; n < 2; ++n) {
    const int f = n*16 + m;
    v16h bL, bH;                                  // B[h][f] = W1[f][h]
    #pragma unroll
    for (int j = 0; j < 16; ++j) {
      bL[j] = (_Float16)((f < 17) ? W1[f*HID + (kbB + j)]      : 0.0f);
      bH[j] = (_Float16)((f < 17) ? W1[f*HID + (32 + kbB + j)] : 0.0f);
    }
    v8f c;
    #pragma unroll
    for (int r = 0; r < 8; ++r) c[r] = 0.0f;
    c = wmma_f16(qL, bL, c);
    c = wmma_f16(qH, bH, c);
    if (f < 17) {
      #pragma unroll
      for (int r = 0; r < 8; ++r)
        gdesc[((size_t)tile*16 + rbase + r)*32 + f] = c[r];
    }
  }
}

// -------------------------------------------------------------------------
// K4: per-atom 3-body backward + final gradient assembly into d_out
// -------------------------------------------------------------------------
__global__ __launch_bounds__(64) void threebody_bwd_kernel(const float* __restrict__ wsA,
                                                           const float* __restrict__ wsB,
                                                           const float* __restrict__ gdesc,
                                                           const float* __restrict__ eij,
                                                           float* __restrict__ out) {
  __shared__ float sdn[KNEI*3];
  __shared__ float sw[KNEI];
  __shared__ float sg[KNEI*12];
  int t = threadIdx.x;
  int atom = blockIdx.x;
  if (t < KNEI) {
    size_t p = (size_t)atom*KNEI + t;
    const float* A = wsA + p*8;
    sdn[t*3+0]=A[1]; sdn[t*3+1]=A[2]; sdn[t*3+2]=A[3];
    sw[t] = A[6];
    const float* G = gdesc + p*32 + 5;
    #pragma unroll
    for (int m = 0; m < 12; ++m) sg[t*12+m] = G[m];
  }
  __syncthreads();
  if (t >= KNEI) return;
  int k = t;
  float gk[12];
  #pragma unroll
  for (int m = 0; m < 12; ++m) gk[m] = sg[k*12+m];
  float dkx = sdn[k*3], dky = sdn[k*3+1], dkz = sdn[k*3+2];
  float wk = sw[k];
  float gx = 0.f, gy = 0.f, gz = 0.f, gfcr = 0.f;
  for (int l = 0; l < KNEI; ++l) {
    float lx = sdn[l*3], ly = sdn[l*3+1], lz = sdn[l*3+2];
    float c = (l == k) ? 0.0f : (dkx*lx + dky*ly + dkz*lz);  // diagonal masked pre-exp
    float wl = sw[l];
    float Skl = 0.f, Slk = 0.f;
    #pragma unroll
    for (int m = 0; m < 12; ++m) {
      float u = c - (-1.0f + 0.18181818181818182f * (float)m);
      float T = __expf(-4.0f * u * u);
      float glm = sg[l*12+m];
      Skl  += gk[m] * T * u;    // dT/dc factor u, scaled later by -2*eta
      Slk  += glm   * T * u;
      gfcr += glm   * T;        // symmetric: g_fcrik_k via c_lk == c_kl (diag included)
    }
    if (l != k) {
      float coef = -8.0f * (wl*Skl + wk*Slk);  // -2*ETA = -8
      gx += coef * lx; gy += coef * ly; gz += coef * lz;
    }
  }
  size_t p = (size_t)atom*KNEI + k;
  const float* A = wsA + p*8;
  float d = A[0], fcv = A[4], dfc = A[5], dfcr = A[7];
  const float* B = wsB + p*12;
  const float* G = gdesc + p*32;
  float sB = 0.f, sBp = 0.f;
  #pragma unroll
  for (int n = 0; n < 5; ++n) { float gr = G[n]; sB += gr*B[n]; sBp += gr*B[5+n]; }
  // dE/dd = fc*Σg·B' + fc'(d)*(Σg·B + eij) + g_fcrik * fcrik'(d)
  float g_d = fcv*sBp + dfc*(sB + eij[p]) + gfcr*dfcr;
  float dot = gx*dkx + gy*dky + gz*dkz;
  float invm = 1.0f / fmaxf(d, 1e-12f);
  out[p*3+0] = g_d*dkx + (gx - dot*dkx) * invm;
  out[p*3+1] = g_d*dky + (gy - dot*dky) * invm;
  out[p*3+2] = g_d*dkz + (gz - dot*dkz) * invm;
}

extern "C" void kernel_launch(void* const* d_in, const int* in_sizes, int n_in,
                              void* d_out, int out_size, void* d_ws, size_t ws_size,
                              hipStream_t stream) {
  (void)in_sizes; (void)n_in; (void)out_size; (void)ws_size;
  const float* rij = (const float*)d_in[0];
  // d_in[1] = unique_i (int64) is implied by the dense layout; unused.
  const float* W1 = (const float*)d_in[2];
  const float* b1 = (const float*)d_in[3];
  const float* W2 = (const float*)d_in[4];
  const float* b2 = (const float*)d_in[5];
  const float* W3 = (const float*)d_in[6];
  const float* b3 = (const float*)d_in[7];
  float* out = (float*)d_out;

  float* wsA   = (float*)d_ws;                    // NPAIRS*8
  float* wsB   = wsA   + (size_t)NPAIRS * 8;      // NPAIRS*12
  float* descw = wsB   + (size_t)NPAIRS * 12;     // NPAIRS*32
  float* gdes  = descw + (size_t)NPAIRS * 32;     // NPAIRS*32
  float* eijw  = gdes  + (size_t)NPAIRS * 32;     // NPAIRS

  pair_prep_kernel<<<(NPAIRS + 255)/256, 256, 0, stream>>>(rij, wsA, wsB, descw);
  threebody_fwd_kernel<<<NATOMS, 64, 0, stream>>>(wsA, descw);
  mlp_fwd_bwd_kernel<<<NPAIRS/16, 32, 0, stream>>>(descw, wsA, W1, b1, W2, b2, W3, b3, gdes, eijw);
  threebody_bwd_kernel<<<NATOMS, 64, 0, stream>>>(wsA, wsB, gdes, eijw, out);
}
